// Attention_18872086298929
// MI455X (gfx1250) — matrix operand order repaired and checked by reading, not compiled, not run on hardware
//
#include <hip/hip_runtime.h>
#include <hip/hip_bf16.h>

// Problem constants (match reference)
#define BB   64
#define TT   2048
#define DH   1024
#define DZ   1024
#define DA   1024

typedef __attribute__((ext_vector_type(16))) __bf16 v16bf;
typedef __attribute__((ext_vector_type(8)))  float  v8f;
typedef __attribute__((ext_vector_type(4)))  float  f4;

// Types matching the async-to-LDS builtin signature (from compiler diagnostic):
// param type is '__attribute__((vector_size(4*sizeof(int)))) int __device__ *'
typedef int b128_t __attribute__((vector_size(16)));
typedef __attribute__((address_space(1))) b128_t* global_b128_p;
typedef __attribute__((address_space(3))) b128_t* lds_b128_p;

#if __has_builtin(__builtin_amdgcn_global_load_async_to_lds_b128) && \
    __has_builtin(__builtin_amdgcn_s_wait_asynccnt)
#define USE_ASYNC_LDS 1
#endif

// Hardware tanh (V_TANH_F32 on gfx1250) with a branch-free fallback.
static __device__ __forceinline__ float fast_tanh(float x) {
#if __has_builtin(__builtin_amdgcn_tanhf)
    return __builtin_amdgcn_tanhf(x);
#elif __has_builtin(__builtin_amdgcn_tanh_f32)
    return __builtin_amdgcn_tanh_f32(x);
#else
    // tanh(x) = 1 - 2/(exp(2x)+1): v_exp + v_rcp, no exec-mask branches
    float e = __expf(2.0f * x);
    return 1.0f - 2.0f * __builtin_amdgcn_rcpf(e + 1.0f);
#endif
}

// ---------------------------------------------------------------------------
// Kernel 1: convert W_h = W[:, 0:DH] (f32, row-major [DA, DH+DZ]) into bf16
// WMMA B-fragment order. B matrix is 32x16 (KxN), B[k][n] = W[a=n][d=k].
// Layout per (nt, kt) block of 512 bf16 (1 KB):
//   lane l<16 : n = nt*16+l,    elements 0..15 = K kt*32+{0..7, 16..23}
//   lane l>=16: n = nt*16+l-16, elements 0..15 = K kt*32+{8..15, 24..31}
// ---------------------------------------------------------------------------
__global__ void __launch_bounds__(256)
prep_w_kernel(const float* __restrict__ W, __bf16* __restrict__ Wf) {
    const int g    = blockIdx.x * 256 + threadIdx.x;
    const int blk  = g >> 5;              // nt*32 + kt
    const int lane = g & 31;
    const int nt   = blk >> 5;
    const int kt   = blk & 31;
    const int n    = nt * 16 + (lane & 15);
    const int k    = kt * 32 + ((lane >> 4) ? 8 : 0);
    const float* src = W + (size_t)n * (DH + DZ) + k;
    v16bf fr;
#pragma unroll
    for (int j = 0; j < 16; ++j) fr[j] = (__bf16)src[j < 8 ? j : j + 8];
    *(v16bf*)(Wf + (size_t)blk * 512 + lane * 16) = fr;
}

// ---------------------------------------------------------------------------
// Kernel 2: u[b][a] = bias[a] + sum_d W[a][DH+d] * Z[b][d]   (tiny: 134 MFLOP)
// ---------------------------------------------------------------------------
__global__ void prep_u_kernel(const float* __restrict__ W, const float* __restrict__ Z,
                              const float* __restrict__ bias, float* __restrict__ U) {
    const int idx = blockIdx.x * 256 + threadIdx.x;  // b*DA + a
    const int b = idx >> 10;
    const int a = idx & (DA - 1);
    const f4* wr = (const f4*)(W + (size_t)a * (DH + DZ) + DH);
    const f4* zr = (const f4*)(Z + (size_t)b * DZ);
    float s = bias[a];
#pragma unroll 4
    for (int d = 0; d < DZ / 4; ++d) {
        f4 wv = wr[d], zv = zr[d];
        s += wv[0] * zv[0] + wv[1] * zv[1] + wv[2] * zv[2] + wv[3] * zv[3];
    }
    U[idx] = s;
}

// ---------------------------------------------------------------------------
// Kernel 3 (main): score[b,t] = sum_a v[a] * tanh( (W_h h)[a] + u[b,a] )
// 4 waves/WG, each wave owns 16 rows of H. A panel (16x1024 bf16) staged once
// in LDS in A-fragment order; B fragments stream from the prepped global
// buffer (L0/L2 resident, 2 MB total) so vmem and ds pipes overlap.
// ---------------------------------------------------------------------------
__global__ void __launch_bounds__(128)
score_kernel(const float* __restrict__ H, const __bf16* __restrict__ Wf,
             const float* __restrict__ U, const float* __restrict__ V,
             float* __restrict__ score) {
    extern __shared__ char smem[];
    const int tid  = threadIdx.x;
    const int lane = tid & 31;
    const int w    = tid >> 5;
    const int r0   = blockIdx.x * 64;        // first global row (b*T + t)
    const int bb   = r0 >> 11;               // / TT
    const int rw   = r0 + w * 16;            // this wave's first row
    char* aw = smem + w * 32768;             // 32 KB per-wave A panel

    // ---- Stage A: 16 rows x 1024 cols f32 -> bf16, ISA A-fragment order ----
    // 16-bit A 16x32: lane<16 (m=lane):  K runs {0..7, 16..23}
    //                 lane>=16(m=lane-16):K runs {8..15, 24..31}
    const int m  = lane & 15;
    const int hi = lane >> 4;
    const float* hrow = H + (size_t)(rw + m) * DH;
#pragma unroll 4
    for (int kt = 0; kt < 32; ++kt) {
        const int k0 = kt * 32 + hi * 8;
        f4 a0 = *(const f4*)(hrow + k0);
        f4 a1 = *(const f4*)(hrow + k0 + 4);
        f4 a2 = *(const f4*)(hrow + k0 + 16);
        f4 a3 = *(const f4*)(hrow + k0 + 20);
        v16bf fr;
        fr[0]  = (__bf16)a0[0]; fr[1]  = (__bf16)a0[1]; fr[2]  = (__bf16)a0[2]; fr[3]  = (__bf16)a0[3];
        fr[4]  = (__bf16)a1[0]; fr[5]  = (__bf16)a1[1]; fr[6]  = (__bf16)a1[2]; fr[7]  = (__bf16)a1[3];
        fr[8]  = (__bf16)a2[0]; fr[9]  = (__bf16)a2[1]; fr[10] = (__bf16)a2[2]; fr[11] = (__bf16)a2[3];
        fr[12] = (__bf16)a3[0]; fr[13] = (__bf16)a3[1]; fr[14] = (__bf16)a3[2]; fr[15] = (__bf16)a3[3];
        *(v16bf*)(aw + kt * 1024 + lane * 32) = fr;
    }
    __syncthreads();

    // ---- Main loop: 64 N-tiles x 32 K-tiles of v_wmma_f32_16x16x32_bf16 ----
    float p[8] = {0, 0, 0, 0, 0, 0, 0, 0};
    const float* ub = U + (size_t)bb * DA;
    for (int nt = 0; nt < 64; ++nt) {
        const int   n  = nt * 16 + m;
        const float un = ub[n];
        const float vn = V[n];
        v8f acc = {0, 0, 0, 0, 0, 0, 0, 0};
        const __bf16* bptr = Wf + (size_t)nt * 32 * 512 + lane * 16;
        if (nt < 63)  // warm next N-tile's fragment stream (global_prefetch_b8)
            __builtin_prefetch(bptr + 32 * 512, 0, 1);
#pragma unroll 8
        for (int kt = 0; kt < 32; ++kt) {
            v16bf av = *(const v16bf*)(aw + kt * 1024 + lane * 32);
            v16bf bv = *(const v16bf*)(bptr + kt * 512);
            acc = __builtin_amdgcn_wmma_f32_16x16x32_bf16(
                false, av, false, bv, (short)0, acc, false, false);
        }
        // Epilogue on C/D layout: VGPR i -> row (hi? i+8 : i), lane&15 -> col
#pragma unroll
        for (int i = 0; i < 8; ++i)
            p[i] += vn * fast_tanh(acc[i] + un);
    }

    // ---- Per-row reduce over the 16 lanes of each half-wave ----
#pragma unroll
    for (int i = 0; i < 8; ++i) {
        float x = p[i];
        x += __shfl_xor(x, 1, 32);
        x += __shfl_xor(x, 2, 32);
        x += __shfl_xor(x, 4, 32);
        x += __shfl_xor(x, 8, 32);
        p[i] = x;
    }
    if (m == 0) {
#pragma unroll
        for (int i = 0; i < 8; ++i)
            score[rw + (hi ? i + 8 : i)] = p[i];
    }
}

// ---------------------------------------------------------------------------
// Kernel 4: softmax over T per batch row
// ---------------------------------------------------------------------------
__global__ void softmax_kernel(const float* __restrict__ score, float* __restrict__ alpha) {
    __shared__ float red[256];
    const int b = blockIdx.x, tid = threadIdx.x;
    const float* s = score + (size_t)b * TT;
    float* a = alpha + (size_t)b * TT;

    float mx = -3.0e38f;
    for (int t = tid; t < TT; t += 256) mx = fmaxf(mx, s[t]);
    red[tid] = mx; __syncthreads();
    for (int off = 128; off; off >>= 1) {
        if (tid < off) red[tid] = fmaxf(red[tid], red[tid + off]);
        __syncthreads();
    }
    mx = red[0]; __syncthreads();

    float sum = 0.0f;
    for (int t = tid; t < TT; t += 256) {
        float e = __expf(s[t] - mx);
        a[t] = e;
        sum += e;
    }
    red[tid] = sum; __syncthreads();
    for (int off = 128; off; off >>= 1) {
        if (tid < off) red[tid] += red[tid + off];
        __syncthreads();
    }
    const float inv = 1.0f / red[0];
    for (int t = tid; t < TT; t += 256) a[t] *= inv;
}

// ---------------------------------------------------------------------------
// Kernel 5: C[b][d] = sum_t alpha[b,t] * H[b,t,d]   (HBM-streaming bound)
// One block per b; 256 threads x float4 cover DH=1024.
// Async path: double-buffered 16-row (64 KB) tiles pulled HBM->LDS with
// global_load_async_to_lds_b128 (ASYNCcnt), copy of tile i+1 in flight while
// tile i is consumed from LDS.
// ---------------------------------------------------------------------------
#define CTX_ROWS 16

__global__ void __launch_bounds__(256)
context_kernel(const float* __restrict__ alpha, const float* __restrict__ H,
               float* __restrict__ out) {
    __shared__ float sa[TT];
    const int b = blockIdx.x, tid = threadIdx.x;
    for (int t = tid; t < TT; t += 256) sa[t] = alpha[(size_t)b * TT + t];

#if defined(USE_ASYNC_LDS)
    __shared__ float htile[2][CTX_ROWS * DH];        // 2 x 64 KB
    const char* hbase = (const char*)(H + (size_t)b * TT * DH);

    // 64 KB tile = 4096 16B chunks; 256 threads x 16 chunks each
    auto issue_tile = [&](int tile, int buf) {
        const size_t src0 = (size_t)(uintptr_t)(hbase + (size_t)tile * CTX_ROWS * DH * 4);
        const unsigned dst0 = (unsigned)(uintptr_t)(&htile[buf][0]);
#pragma unroll
        for (int i = 0; i < 16; ++i) {
            const int c = tid + i * 256;
            __builtin_amdgcn_global_load_async_to_lds_b128(
                (global_b128_p)(uintptr_t)(src0 + (size_t)c * 16),
                (lds_b128_p)(uintptr_t)(dst0 + c * 16),
                0, 0);
        }
    };

    issue_tile(0, 0);
    __builtin_amdgcn_s_wait_asynccnt(0);
    __syncthreads();                                  // tile 0 + sa[] visible

    f4 acc = {0, 0, 0, 0};
    for (int tile = 0; tile < TT / CTX_ROWS; ++tile) {
        const int buf = tile & 1;
        if (tile + 1 < TT / CTX_ROWS) issue_tile(tile + 1, buf ^ 1);
        const float* tp = &htile[buf][0];
#pragma unroll
        for (int j = 0; j < CTX_ROWS; ++j) {
            const float av = sa[tile * CTX_ROWS + j];
            const f4 hv = *(const f4*)(tp + j * DH + tid * 4);
            acc[0] += av * hv[0];
            acc[1] += av * hv[1];
            acc[2] += av * hv[2];
            acc[3] += av * hv[3];
        }
        __builtin_amdgcn_s_wait_asynccnt(0);          // next tile landed
        __syncthreads();                              // all waves done with buf
    }
    ((f4*)out)[(size_t)b * (DH / 4) + tid] = acc;
#else
    __syncthreads();
    const f4* h4 = (const f4*)(H + (size_t)b * TT * DH);
    f4 acc = {0, 0, 0, 0};
#pragma unroll 4
    for (int t = 0; t < TT; ++t) {
        if (t + 16 < TT)
            __builtin_prefetch(&h4[(size_t)(t + 16) * (DH / 4) + tid], 0, 1);
        f4 hv = h4[(size_t)t * (DH / 4) + tid];
        const float av = sa[t];
        acc[0] += av * hv[0];
        acc[1] += av * hv[1];
        acc[2] += av * hv[2];
        acc[3] += av * hv[3];
    }
    ((f4*)out)[(size_t)b * (DH / 4) + tid] = acc;
#endif
}

// ---------------------------------------------------------------------------
// Launch
// ---------------------------------------------------------------------------
extern "C" void kernel_launch(void* const* d_in, const int* in_sizes, int n_in,
                              void* d_out, int out_size, void* d_ws, size_t ws_size,
                              hipStream_t stream) {
    const float* Z    = (const float*)d_in[0];  // [B, DZ]
    const float* H    = (const float*)d_in[1];  // [B, T, DH]
    const float* W    = (const float*)d_in[2];  // [DA, DH+DZ]
    const float* bias = (const float*)d_in[3];  // [DA]
    const float* V    = (const float*)d_in[4];  // [DA]
    float* out = (float*)d_out;                 // [B, DH]

    char* ws = (char*)d_ws;
    __bf16* Wf    = (__bf16*)ws;                                  // 2 MB bf16 frags
    float*  U     = (float*)(ws + (2u << 20));                    // 256 KB
    float*  score = (float*)(ws + (2u << 20) + (256u << 10));     // 512 KB
    float*  alpha = (float*)(ws + (2u << 20) + (768u << 10));     // 512 KB

    prep_w_kernel<<<(DA / 16) * (DH / 32) / 8, 256, 0, stream>>>(W, Wf);
    prep_u_kernel<<<(BB * DA) / 256, 256, 0, stream>>>(W, Z, bias, U);
    score_kernel<<<(BB * TT) / 64, 128, 4 * 32768, stream>>>(H, Wf, U, V, score);
    softmax_kernel<<<BB, 256, 0, stream>>>(score, alpha);
    context_kernel<<<BB, 256, 0, stream>>>(alpha, H, out);
}
